// Model_36532991820041
// MI455X (gfx1250) — matrix operand order repaired and checked by reading, not compiled, not run on hardware
//
#include <hip/hip_runtime.h>

// ---------------------------------------------------------------------------
// Types for CDNA5 WMMA
// ---------------------------------------------------------------------------
typedef __bf16 v16bf __attribute__((ext_vector_type(16)));
typedef __bf16 v8bf  __attribute__((ext_vector_type(8)));
typedef float  v8f   __attribute__((ext_vector_type(8)));

#define NNODES 50000
#define NEDGES 800000
#define DIN    47
#define HID    1024

#define TILE_M 128
#define TILE_N 128
#define TILE_K 32

// ---------------------------------------------------------------------------
// CDNA5 async global->LDS copy (ASYNCcnt-tracked, bypasses VGPR staging)
// ---------------------------------------------------------------------------
__device__ __forceinline__ void async_copy_b128(unsigned lds_off, const void* g) {
    unsigned long long ga = (unsigned long long)g;
    asm volatile("global_load_async_to_lds_b128 %0, %1, off"
                 :: "v"(lds_off), "v"(ga) : "memory");
}

__device__ __forceinline__ void wait_asynccnt0() {
#if __has_builtin(__builtin_amdgcn_s_wait_asynccnt)
    __builtin_amdgcn_s_wait_asynccnt(0);
#else
    asm volatile("s_wait_asynccnt 0x0" ::: "memory");
#endif
}

// ---------------------------------------------------------------------------
// Utility kernels
// ---------------------------------------------------------------------------
__global__ void zero_f32(float* __restrict__ p, int n) {
    int id = blockIdx.x * blockDim.x + threadIdx.x;
    if (id < n) p[id] = 0.0f;
}

// Build transposed, concatenated bf16 weights: Wt[n][k], k < H -> Wl, else Wr.
// H = padded half-K (64 for layer 1, 1024 for layers 2/3); valid = real rows.
__global__ void cast_weights(const float* __restrict__ Wl,
                             const float* __restrict__ Wr,
                             __bf16* __restrict__ Wt, int H, int valid) {
    int id = blockIdx.x * blockDim.x + threadIdx.x;
    int Kc = 2 * H;
    if (id >= HID * Kc) return;
    int n = id / Kc;
    int k = id - n * Kc;
    int kk = (k < H) ? k : (k - H);
    const float* W = (k < H) ? Wl : Wr;
    float v = (kk < valid) ? W[(size_t)kk * HID + n] : 0.0f;
    Wt[(size_t)n * Kc + k] = (__bf16)v;
}

// Fill self-half of layer-1 activations: A1[n][64+k] = bf16(x[n][k]), pad 0.
__global__ void fill_a1_self(const float* __restrict__ x,
                             __bf16* __restrict__ A1, int N) {
    int id = blockIdx.x * blockDim.x + threadIdx.x;
    if (id >= N * 64) return;
    int n = id >> 6, k = id & 63;
    float v = (k < DIN) ? x[(size_t)n * DIN + k] : 0.0f;
    A1[(size_t)n * 128 + 64 + k] = (__bf16)v;
}

__global__ void degree_kernel(const int* __restrict__ dst,
                              float* __restrict__ deg, int E) {
    int id = blockIdx.x * blockDim.x + threadIdx.x;
    if (id < E) atomicAdd(&deg[dst[id]], 1.0f);
}

// Layer-1 scatter: accum[d][k] += x[s][k] (fp32 features, feat = 47)
__global__ void scatter_f32(const float* __restrict__ x,
                            const int* __restrict__ src,
                            const int* __restrict__ dst,
                            float* __restrict__ accum, int feat) {
    int e = blockIdx.x;
    int s = src[e], d = dst[e];
    for (int k = threadIdx.x; k < feat; k += blockDim.x)
        atomicAdd(&accum[(size_t)d * feat + k], x[(size_t)s * feat + k]);
}

// Layers 2/3 scatter: gather bf16 self-half of previous layer, f32 atomic add.
__global__ void scatter_bf16(const __bf16* __restrict__ h,
                             const int* __restrict__ src,
                             const int* __restrict__ dst,
                             float* __restrict__ accum,
                             int feat, int stride, int off) {
    int e = blockIdx.x;
    int s = src[e], d = dst[e];
    const __bf16* hs = h + (size_t)s * stride + off;
    for (int k = threadIdx.x; k < feat; k += blockDim.x)
        atomicAdd(&accum[(size_t)d * feat + k], (float)hs[k]);
}

// Normalize layer-1 mean into A1[n][0..63] (bf16, zero-padded past 47).
__global__ void norm_mean1(const float* __restrict__ accum,
                           const float* __restrict__ deg,
                           __bf16* __restrict__ A1, int N) {
    int id = blockIdx.x * blockDim.x + threadIdx.x;
    if (id >= N * 64) return;
    int n = id >> 6, k = id & 63;
    float dg = deg[n];
    dg = dg < 1.0f ? 1.0f : dg;
    float v = (k < DIN) ? accum[(size_t)n * DIN + k] / dg : 0.0f;
    A1[(size_t)n * 128 + k] = (__bf16)v;
}

// Normalize 1024-wide mean into mean-half of a [N,2048] activation buffer.
__global__ void norm_mean(const float* __restrict__ accum,
                          const float* __restrict__ deg,
                          __bf16* __restrict__ A, int N) {
    int id = blockIdx.x * blockDim.x + threadIdx.x;
    if (id >= N * HID) return;
    int n = id >> 10, k = id & 1023;
    float dg = deg[n];
    dg = dg < 1.0f ? 1.0f : dg;
    A[(size_t)n * 2048 + k] = (__bf16)(accum[(size_t)n * HID + k] / dg);
}

// ---------------------------------------------------------------------------
// WMMA GEMM: out = relu(A[n_rows,Kc] @ Wt^T + bias), bf16 in, f32 acc, bf16 out.
// Wt is [1024, Kc] (output-column major) so B tiles load contiguously.
// Block = 256 threads = 8 wave32s; block tile 128x128, K-step 32.
// Double-buffered LDS fed by async global->LDS copies (ASYNCcnt), one
// barrier per K-step. All fragments loaded to registers before the WMMA
// burst so ds_load latency is pipelined against independent WMMAs.
// ---------------------------------------------------------------------------
__global__ __launch_bounds__(256)
void gemm_bias_relu_bf16(const __bf16* __restrict__ A,
                         const __bf16* __restrict__ Wt,
                         const float* __restrict__ bias,
                         __bf16* __restrict__ out,
                         int n_rows, int Kc, int out_stride, int out_off) {
    __shared__ __align__(16) __bf16 Alds[2][TILE_M * TILE_K];
    __shared__ __align__(16) __bf16 Blds[2][TILE_N * TILE_K];

    const int t    = threadIdx.x;
    const int wave = t >> 5;        // 0..7 (wave32)
    const int lane = t & 31;
    const int r    = lane & 15;
    const int hi   = lane >> 4;
    const int m0   = blockIdx.x * TILE_M;
    const int n0   = blockIdx.y * TILE_N;

    v8f acc[8];
#pragma unroll
    for (int i = 0; i < 8; ++i) { v8f z = {}; acc[i] = z; }

    // Cooperative tile loaders: thread -> (row 0..127, 16-elem half)
    const int lrow  = t >> 1;
    const int lhalf = (t & 1) * 16;
    int arow = m0 + lrow;
    if (arow >= n_rows) arow = n_rows - 1;   // clamp (stores are guarded)
    const __bf16* Agp = A  + (size_t)arow * Kc + lhalf;
    const __bf16* Bgp = Wt + (size_t)(n0 + lrow) * Kc + lhalf;

    // Wave-relative LDS byte offsets for this thread's 32B slot in each buffer
    // (low 32 bits of a generic pointer to a __shared__ object are the LDS
    // offset per the flat-aperture mapping).
    const unsigned lslot = (unsigned)((lrow * TILE_K + lhalf) * sizeof(__bf16));
    const unsigned aL0 = (unsigned)(uintptr_t)(&Alds[0][0]) + lslot;
    const unsigned aL1 = (unsigned)(uintptr_t)(&Alds[1][0]) + lslot;
    const unsigned bL0 = (unsigned)(uintptr_t)(&Blds[0][0]) + lslot;
    const unsigned bL1 = (unsigned)(uintptr_t)(&Blds[1][0]) + lslot;

    auto issue = [&](int buf, int k0) {
        const __bf16* ga = Agp + k0;
        const __bf16* gb = Bgp + k0;
        const unsigned aL = buf ? aL1 : aL0;
        const unsigned bL = buf ? bL1 : bL0;
        async_copy_b128(aL,      ga);
        async_copy_b128(aL + 16, ga + 8);
        async_copy_b128(bL,      gb);
        async_copy_b128(bL + 16, gb + 8);
    };

    issue(0, 0);                       // prologue: buffer 0 <- k0 = 0
    int buf = 0;

    for (int k0 = 0; k0 < Kc; k0 += TILE_K) {
        wait_asynccnt0();              // this wave's async writes landed in LDS
        __syncthreads();               // everyone's writes landed; prev reads done

        if (k0 + TILE_K < Kc) issue(buf ^ 1, k0 + TILE_K);

        const v8bf* Al8 = reinterpret_cast<const v8bf*>(&Alds[buf][0]);
        const v8bf* Bl8 = reinterpret_cast<const v8bf*>(&Blds[buf][0]);

        // ---- Load ALL fragments into registers first (batched ds_loads) ----
        // A fragment (16x32 bf16): K in [8hi,8hi+8) then [16+8hi,24+8hi)
        const int abase = (wave * 16 + r) * TILE_K;
        v8bf a_lo = Al8[(abase + 8 * hi) >> 3];
        v8bf a_hi = Al8[(abase + 16 + 8 * hi) >> 3];
        v16bf afrag = __builtin_shufflevector(
            a_lo, a_hi, 0, 1, 2, 3, 4, 5, 6, 7, 8, 9, 10, 11, 12, 13, 14, 15);

        v16bf bfrag[8];
#pragma unroll
        for (int nt = 0; nt < 8; ++nt) {
            // B fragment (32x16 bf16): col = nt*16+r, K in [8hi,8hi+8) then [16+8hi,24+8hi)
            const int bbase = (nt * 16 + r) * TILE_K;
            v8bf b_lo = Bl8[(bbase + 8 * hi) >> 3];
            v8bf b_hi = Bl8[(bbase + 16 + 8 * hi) >> 3];
            bfrag[nt] = __builtin_shufflevector(
                b_lo, b_hi, 0, 1, 2, 3, 4, 5, 6, 7, 8, 9, 10, 11, 12, 13, 14, 15);
        }

        // ---- WMMA burst: 8 independent accumulators, shared A operand ----
#pragma unroll
        for (int nt = 0; nt < 8; ++nt) {
            acc[nt] = __builtin_amdgcn_wmma_f32_16x16x32_bf16(
                false, afrag, false, bfrag[nt], (short)0, acc[nt], false, false);
        }
        buf ^= 1;
    }

    // Store: lane (r=N, hi) owns rows m_base + e + 8*hi for VGPR e.
#pragma unroll
    for (int nt = 0; nt < 8; ++nt) {
        const int col = n0 + nt * 16 + r;
        const float bv = bias[col];
#pragma unroll
        for (int e = 0; e < 8; ++e) {
            const int row = m0 + wave * 16 + e + 8 * hi;
            if (row < n_rows) {
                float v = acc[nt][e] + bv;
                v = v > 0.0f ? v : 0.0f;
                out[(size_t)row * out_stride + out_off + col] = (__bf16)v;
            }
        }
    }
}

// Final head: out[n] = sum_k h[n,k]*w[k] + b. One wave32 per node.
__global__ __launch_bounds__(256)
void gemv_final(const __bf16* __restrict__ h, const float* __restrict__ w,
                const float* __restrict__ b, float* __restrict__ out, int N) {
    const int wave = threadIdx.x >> 5;
    const int lane = threadIdx.x & 31;
    const int n = blockIdx.x * 8 + wave;
    if (n >= N) return;
    const __bf16* hn = h + (size_t)n * HID;
    float s = 0.0f;
    for (int k = lane; k < HID; k += 32) s += (float)hn[k] * w[k];
#pragma unroll
    for (int off = 16; off > 0; off >>= 1) s += __shfl_xor(s, off, 32);
    if (lane == 0) out[n] = s + b[0];
}

// ---------------------------------------------------------------------------
// Launch
// ---------------------------------------------------------------------------
extern "C" void kernel_launch(void* const* d_in, const int* in_sizes, int n_in,
                              void* d_out, int out_size, void* d_ws, size_t ws_size,
                              hipStream_t stream) {
    const int N = NNODES, E = NEDGES;

    const float* feat = (const float*)d_in[0];
    const int*   ei   = (const int*)d_in[1];
    const int*   src  = ei;
    const int*   dst  = ei + E;
    const float* W1l  = (const float*)d_in[2];
    const float* b1   = (const float*)d_in[3];
    const float* W1r  = (const float*)d_in[4];
    const float* W2l  = (const float*)d_in[5];
    const float* b2   = (const float*)d_in[6];
    const float* W2r  = (const float*)d_in[7];
    const float* W3l  = (const float*)d_in[8];
    const float* b3   = (const float*)d_in[9];
    const float* W3r  = (const float*)d_in[10];
    const float* Wlin = (const float*)d_in[11];
    const float* blin = (const float*)d_in[12];

    // Workspace layout (256B aligned)
    char* ws = (char*)d_ws;
    size_t off = 0;
    auto alloc = [&](size_t bytes) {
        char* p = ws + off;
        off = (off + bytes + 255) & ~(size_t)255;
        return p;
    };
    float*  accum = (float*)alloc((size_t)N * HID * 4);   // 204.8 MB, reused
    __bf16* A2    = (__bf16*)alloc((size_t)N * 2048 * 2); // [mean|self] layer2 in
    __bf16* A3    = (__bf16*)alloc((size_t)N * 2048 * 2); // [mean|self] layer3 in
    __bf16* A1    = (__bf16*)alloc((size_t)N * 128 * 2);  // [mean64|self64]
    float*  deg   = (float*)alloc((size_t)N * 4);
    __bf16* Wc1   = (__bf16*)alloc((size_t)HID * 128 * 2);
    __bf16* Wc2   = (__bf16*)alloc((size_t)HID * 2048 * 2);
    __bf16* Wc3   = (__bf16*)alloc((size_t)HID * 2048 * 2);
    __bf16* h3    = A2;  // A2 dead once layer-3 GEMM reads A3

    const int B = 256;
    const dim3 ggrid((N + TILE_M - 1) / TILE_M, HID / TILE_N);

    // Graph-invariant prep (recomputed every call: no hidden state)
    zero_f32<<<(N + B - 1) / B, B, 0, stream>>>(deg, N);
    zero_f32<<<(N * DIN + B - 1) / B, B, 0, stream>>>(accum, N * DIN);
    cast_weights<<<(HID * 128 + B - 1) / B, B, 0, stream>>>(W1l, W1r, Wc1, 64, DIN);
    cast_weights<<<(HID * 2048 + B - 1) / B, B, 0, stream>>>(W2l, W2r, Wc2, HID, HID);
    cast_weights<<<(HID * 2048 + B - 1) / B, B, 0, stream>>>(W3l, W3r, Wc3, HID, HID);
    fill_a1_self<<<(N * 64 + B - 1) / B, B, 0, stream>>>(feat, A1, N);
    degree_kernel<<<(E + B - 1) / B, B, 0, stream>>>(dst, deg, E);

    // Layer 1
    scatter_f32<<<E, 64, 0, stream>>>(feat, src, dst, accum, DIN);
    norm_mean1<<<(N * 64 + B - 1) / B, B, 0, stream>>>(accum, deg, A1, N);
    gemm_bias_relu_bf16<<<ggrid, B, 0, stream>>>(A1, Wc1, b1, A2, N, 128, 2048, 1024);

    // Layer 2
    zero_f32<<<(N * HID + B - 1) / B, B, 0, stream>>>(accum, N * HID);
    scatter_bf16<<<E, B, 0, stream>>>(A2, src, dst, accum, HID, 2048, 1024);
    norm_mean<<<(N * HID + B - 1) / B, B, 0, stream>>>(accum, deg, A2, N);
    gemm_bias_relu_bf16<<<ggrid, B, 0, stream>>>(A2, Wc2, b2, A3, N, 2048, 2048, 1024);

    // Layer 3
    zero_f32<<<(N * HID + B - 1) / B, B, 0, stream>>>(accum, N * HID);
    scatter_bf16<<<E, B, 0, stream>>>(A3, src, dst, accum, HID, 2048, 1024);
    norm_mean<<<(N * HID + B - 1) / B, B, 0, stream>>>(accum, deg, A3, N);
    gemm_bias_relu_bf16<<<ggrid, B, 0, stream>>>(A3, Wc3, b3, h3, N, 2048, 1024, 0);

    // Head
    gemv_final<<<(N + 7) / 8, B, 0, stream>>>(h3, Wlin, blin, (float*)d_out, N);
}
